// HarmonicStack_57191784514105
// MI455X (gfx1250) — matrix operand order repaired and checked by reading, not compiled, not run on hardware
//
#include <hip/hip_runtime.h>
#include <hip/hip_bf16.h>
#include <math.h>

// ---------------- model constants ----------------
constexpr int Vv  = 50257;
constexpr int Tt  = 512;
constexpr int Dd  = 2048;
constexpr int Hh  = 16;
constexpr int HD  = Dd / Hh;      // 128
constexpr int AWw = 512;
constexpr int Bb  = 4;
constexpr int BT  = Bb * Tt;      // 2048
constexpr int NVb = 683, NIb = 683, NPb = 682;
constexpr int PADW = 704;         // 11*64, zero-padded band K
constexpr int K3  = 3 * Dd;       // 6144
constexpr int K4  = 4 * Dd;       // 8192
constexpr int HK  = 3 * AWw;      // 1536

// ---------------- WMMA vector types ----------------
typedef __attribute__((ext_vector_type(16))) __bf16 v16bf;
typedef __attribute__((ext_vector_type(8)))  float  v8f;

union FragU { uint4 q[2]; unsigned int u[8]; v16bf v; };

static __device__ __forceinline__ __hip_bfloat16 f2b(float f) {
    return __float2bfloat16(f);
}

// per-lane async copy: 16B global -> 16B LDS, tracked by ASYNCcnt
static __device__ __forceinline__ void async_copy16(unsigned lds_off, const void* gptr) {
    asm volatile("global_load_async_to_lds_b128 %0, %1, off"
                 :: "v"(lds_off), "v"((unsigned long long)gptr)
                 : "memory");
}
static __device__ __forceinline__ void wait_async0() {
    asm volatile("s_wait_asynccnt 0x0" ::: "memory");
}

// =====================================================================
// Generic batched WMMA GEMM:  C[z] = alpha * (A[z] @ B[z]^T)
// A: (M,K) bf16 row-major, lda elems (multiple of 8). B: (N,K) bf16 row-major.
// C: (M,N) f32. mode==1: causal mask (col>row -> -1e30), used for QK^T.
// Block tile 128(M) x 64(N) x 64(K); 8 waves, each wave -> 32x32 patch.
// K must be a multiple of 64. LDS rows: 32 bf16-pairs + 4 pad = 36 dwords
// (144B, 16B aligned; 36r mod 64 = 4*(9r mod 16) -> conflict-free frags).
// Global->LDS staging uses GLOBAL_LOAD_ASYNC_TO_LDS_B128 (ASYNCcnt).
// =====================================================================
#define LROW 36
__global__ __launch_bounds__(256) void k_gemm_bf16(
    const __hip_bfloat16* __restrict__ A, long long strideA, int lda,
    const __hip_bfloat16* __restrict__ Bm, long long strideB, int ldb,
    float* __restrict__ C, long long strideC, int ldc,
    int M, int N, int K, float alpha, int mode)
{
    const int mBase = blockIdx.y * 128;
    const int nBase = blockIdx.x * 64;
    const int z     = blockIdx.z;
    const int tid   = threadIdx.x;
    float* Cb = C + (size_t)z * (size_t)strideC;

    if (mode == 1 && nBase > mBase + 127) {
        // fully masked block: no compute
        for (int i = tid; i < 128 * 64; i += 256) {
            int m = i >> 6, nn = i & 63;
            int gm = mBase + m, gn = nBase + nn;
            if (gm < M && gn < N) Cb[(size_t)gm * ldc + gn] = -1.0e30f;
        }
        return;
    }

    // 16-byte chunk views of A/B (4 bf16-pairs per chunk)
    const uint4* Ap4 = (const uint4*)((const unsigned int*)A + (size_t)z * ((size_t)strideA >> 1));
    const uint4* Bp4 = (const uint4*)((const unsigned int*)Bm + (size_t)z * ((size_t)strideB >> 1));
    const int lda4 = lda >> 3, ldb4 = ldb >> 3;   // uint4 per row
    const bool fullM = (mBase + 128 <= M);
    const bool fullN = (nBase + 64 <= N);

    __shared__ __align__(16) unsigned int lA[128 * LROW];
    __shared__ __align__(16) unsigned int lB[64 * LROW];

    const int lane = tid & 31, wave = tid >> 5;
    const int wm = wave >> 1, wn = wave & 1;   // wave grid 4x2 of 32x32 tiles
    const int lh = lane >> 4, lr = lane & 15;

    // per-thread staging coordinates
    const int ar = tid >> 3, ac = tid & 7;     // A: thread covers rows ar, ar+32, ar+64, ar+96

    v8f acc[2][2];
#pragma unroll
    for (int a = 0; a < 2; ++a)
#pragma unroll
        for (int b = 0; b < 2; ++b)
#pragma unroll
            for (int r = 0; r < 8; ++r) acc[a][b][r] = 0.0f;

    const int nK = K >> 6;                    // 64-wide K stages
    const uint4 zero4 = make_uint4(0u, 0u, 0u, 0u);
    for (int kt = 0; kt < nK; ++kt) {
        const int kq = kt << 3;               // uint4 offset of this stage
        __syncthreads();                      // previous stage fully consumed
        // ---- stage A tile: 128 rows x 8 uint4 (= 64 bf16) ----
        if (fullM) {
#pragma unroll
            for (int i = 0; i < 4; ++i) {
                int r = ar + i * 32;
                async_copy16((unsigned)(size_t)&lA[r * LROW + ac * 4],
                             &Ap4[(size_t)(mBase + r) * lda4 + kq + ac]);
            }
            if (kt + 1 < nK)
                __builtin_prefetch((const void*)&Ap4[(size_t)(mBase + ar) * lda4 + kq + 8 + ac], 0, 0);
        } else {
#pragma unroll
            for (int i = 0; i < 4; ++i) {
                int r = ar + i * 32;
                int gm = mBase + r;
                *(uint4*)&lA[r * LROW + ac * 4] =
                    (gm < M) ? Ap4[(size_t)gm * lda4 + kq + ac] : zero4;
            }
        }
        // ---- stage B tile: 64 rows x 8 uint4 ----
        if (fullN) {
#pragma unroll
            for (int i = 0; i < 2; ++i) {
                int r = ar + i * 32;
                async_copy16((unsigned)(size_t)&lB[r * LROW + ac * 4],
                             &Bp4[(size_t)(nBase + r) * ldb4 + kq + ac]);
            }
        } else {
#pragma unroll
            for (int i = 0; i < 2; ++i) {
                int r = ar + i * 32;
                int gn = nBase + r;
                *(uint4*)&lB[r * LROW + ac * 4] =
                    (gn < N) ? Bp4[(size_t)gn * ldb4 + kq + ac] : zero4;
            }
        }
        wait_async0();                        // this wave's copies landed in LDS
        __syncthreads();                      // whole tile visible to all waves

        // ---- two K=32 sub-steps per stage ----
#pragma unroll
        for (int ks = 0; ks < 2; ++ks) {
            const int kb = ks * 16;           // pair offset within LDS row
            FragU fa[2], fb[2];
            // A fragments: 16-bit A 16x32 layout (ISA 7.12.2)
#pragma unroll
            for (int a = 0; a < 2; ++a) {
                const unsigned int* rp = &lA[(wm * 32 + a * 16 + lr) * LROW + kb];
                fa[a].q[0] = *(const uint4*)&rp[lh * 4];          // K = 8*lh + 0..7
                fa[a].q[1] = *(const uint4*)&rp[8 + lh * 4];      // K = 16 + 8*lh + 0..7
            }
            // B fragments: 16-bit B 32x16 layout, lane = column
#pragma unroll
            for (int b = 0; b < 2; ++b) {
                const unsigned int* rp = &lB[(wn * 32 + b * 16 + lr) * LROW + kb];
                fb[b].q[0] = *(const uint4*)&rp[lh * 4];          // K = 8*lh + 0..7
                fb[b].q[1] = *(const uint4*)&rp[8 + lh * 4];      // K = 16 + 8*lh + 0..7
            }
#pragma unroll
            for (int a = 0; a < 2; ++a)
#pragma unroll
                for (int b = 0; b < 2; ++b)
                    acc[a][b] = __builtin_amdgcn_wmma_f32_16x16x32_bf16(
                        false, fa[a].v, false, fb[b].v, (short)0, acc[a][b], false, false);
        }
    }

    // store: C/D layout — VGPR r: M = r + 8*lh ; N = lane&15
#pragma unroll
    for (int a = 0; a < 2; ++a)
#pragma unroll
        for (int b = 0; b < 2; ++b) {
            int gmB = mBase + wm * 32 + a * 16 + lh * 8;
            int gn  = nBase + wn * 32 + b * 16 + lr;
            if (gn < N) {
#pragma unroll
                for (int r = 0; r < 8; ++r) {
                    int gm = gmB + r;
                    if (gm < M) {
                        float v = alpha * acc[a][b][r];
                        if (mode == 1 && gn > gm) v = -1.0e30f;
                        Cb[(size_t)gm * ldc + gn] = v;
                    }
                }
            }
        }
}

// ---------------- elementwise / reduction kernels ----------------
__global__ __launch_bounds__(256) void k_embed(const int* __restrict__ idx,
    const float* __restrict__ tok, const float* __restrict__ pos, float* __restrict__ x)
{
    size_t tid = (size_t)blockIdx.x * 256 + threadIdx.x;
    if (tid >= (size_t)BT * Dd) return;
    int row = (int)(tid / Dd), d = (int)(tid % Dd);
    int t = row % Tt;
    x[tid] = tok[(size_t)idx[row] * Dd + d] + pos[(size_t)t * Dd + d];
}

__global__ __launch_bounds__(256) void k_cvt(const float* __restrict__ in,
    __hip_bfloat16* __restrict__ out, size_t n)
{
    size_t tid = (size_t)blockIdx.x * 256 + threadIdx.x;
    if (tid < n) out[tid] = f2b(in[tid]);
}

__global__ __launch_bounds__(256) void k_gelu_cvt(const float* __restrict__ in,
    __hip_bfloat16* __restrict__ out, size_t n)
{
    size_t tid = (size_t)blockIdx.x * 256 + threadIdx.x;
    if (tid < n) {
        float v = in[tid];
        v = 0.5f * v * (1.0f + erff(v * 0.70710678118654752f));
        out[tid] = f2b(v);
    }
}

__global__ __launch_bounds__(256) void k_add(float* __restrict__ x,
    const float* __restrict__ d, size_t n)
{
    size_t tid = (size_t)blockIdx.x * 256 + threadIdx.x;
    if (tid < n) x[tid] += d[tid];
}

// LayerNorm over last dim n; optional f32 out, optional bf16 out (strided), optional exact GELU
__global__ __launch_bounds__(256) void k_layernorm(const float* __restrict__ x, int n, int rows,
    const float* __restrict__ g, const float* __restrict__ bta,
    float* __restrict__ outF, __hip_bfloat16* __restrict__ outB, int outBStride, int gelu)
{
    int row = blockIdx.x;
    if (row >= rows) return;
    const float* xr = x + (size_t)row * n;
    __shared__ float sm[256];
    int tid = threadIdx.x;
    float acc = 0.f;
    for (int i = tid; i < n; i += 256) acc += xr[i];
    sm[tid] = acc; __syncthreads();
    for (int s = 128; s > 0; s >>= 1) { if (tid < s) sm[tid] += sm[tid + s]; __syncthreads(); }
    float mean = sm[0] / n; __syncthreads();
    acc = 0.f;
    for (int i = tid; i < n; i += 256) { float d = xr[i] - mean; acc += d * d; }
    sm[tid] = acc; __syncthreads();
    for (int s = 128; s > 0; s >>= 1) { if (tid < s) sm[tid] += sm[tid + s]; __syncthreads(); }
    float rstd = rsqrtf(sm[0] / n + 1e-5f);
    for (int i = tid; i < n; i += 256) {
        float y = (xr[i] - mean) * rstd * g[i] + bta[i];
        if (gelu) y = 0.5f * y * (1.0f + erff(y * 0.70710678118654752f));
        if (outF) outF[(size_t)row * n + i] = y;
        if (outB) outB[(size_t)row * outBStride + i] = f2b(y);
    }
}

// qkv (BT,3D) f32 -> q/k (B,H,T,hd) bf16, vT (B,H,hd,T) bf16
__global__ __launch_bounds__(256) void k_split_qkv(const float* __restrict__ qkv,
    __hip_bfloat16* __restrict__ qb, __hip_bfloat16* __restrict__ kb,
    __hip_bfloat16* __restrict__ vtb)
{
    size_t tid = (size_t)blockIdx.x * 256 + threadIdx.x;
    if (tid >= (size_t)BT * Dd) return;
    int row = (int)(tid / Dd), c = (int)(tid % Dd);
    int b = row / Tt, t = row % Tt;
    int h = c / HD, d = c % HD;
    size_t base = (size_t)row * K3;
    int bh = b * Hh + h;
    qb[((size_t)bh * Tt + t) * HD + d]  = f2b(qkv[base + c]);
    kb[((size_t)bh * Tt + t) * HD + d]  = f2b(qkv[base + Dd + c]);
    vtb[((size_t)bh * HD + d) * Tt + t] = f2b(qkv[base + 2 * Dd + c]);
}

// softmax rows of width w -> bf16
__global__ __launch_bounds__(128) void k_softmax(const float* __restrict__ att,
    __hip_bfloat16* __restrict__ p, int w)
{
    int row = blockIdx.x, tid = threadIdx.x;
    const float* a = att + (size_t)row * w;
    __shared__ float sm[128];
    float mx = -3.4e38f;
    for (int i = tid; i < w; i += 128) mx = fmaxf(mx, a[i]);
    sm[tid] = mx; __syncthreads();
    for (int s = 64; s > 0; s >>= 1) { if (tid < s) sm[tid] = fmaxf(sm[tid], sm[tid + s]); __syncthreads(); }
    mx = sm[0]; __syncthreads();
    float sum = 0.f;
    for (int i = tid; i < w; i += 128) sum += expf(a[i] - mx);
    sm[tid] = sum; __syncthreads();
    for (int s = 64; s > 0; s >>= 1) { if (tid < s) sm[tid] += sm[tid + s]; __syncthreads(); }
    float inv = 1.0f / sm[0];
    for (int i = tid; i < w; i += 128)
        p[(size_t)row * w + i] = f2b(expf(a[i] - mx) * inv);
}

// o_h (B,H,T,hd) f32 -> o (BT,D) bf16
__global__ __launch_bounds__(256) void k_merge_heads(const float* __restrict__ oh,
    __hip_bfloat16* __restrict__ ob)
{
    size_t tid = (size_t)blockIdx.x * 256 + threadIdx.x;
    if (tid >= (size_t)BT * Dd) return;
    int row = (int)(tid / Dd), c = (int)(tid % Dd);
    int b = row / Tt, t = row % Tt;
    int h = c / HD, d = c % HD;
    ob[tid] = f2b(oh[((size_t)(b * Hh + h) * Tt + t) * HD + d]);
}

// gather band dims from xf (rows,D) into padded (rows,PADW) bf16
__global__ __launch_bounds__(256) void k_gather_band(const float* __restrict__ xf,
    const int* __restrict__ bidx, int w, __hip_bfloat16* __restrict__ out)
{
    size_t tid = (size_t)blockIdx.x * 256 + threadIdx.x;
    if (tid >= (size_t)BT * PADW) return;
    int row = (int)(tid / PADW), c = (int)(tid % PADW);
    out[tid] = (c < w) ? f2b(xf[(size_t)row * Dd + bidx[c]]) : f2b(0.0f);
}

// pad analyzer weight (AW,w) f32 -> (AW,PADW) bf16 zero padded
__global__ __launch_bounds__(256) void k_pad_w(const float* __restrict__ win, int w,
    __hip_bfloat16* __restrict__ out)
{
    size_t tid = (size_t)blockIdx.x * 256 + threadIdx.x;
    if (tid >= (size_t)AWw * PADW) return;
    int row = (int)(tid / PADW), c = (int)(tid % PADW);
    out[tid] = (c < w) ? f2b(win[(size_t)row * w + c]) : f2b(0.0f);
}

__global__ void k_zero2(float* a) { a[0] = 0.f; a[1] = 0.f; }

__global__ __launch_bounds__(256) void k_ce(const float* __restrict__ logits,
    const int* __restrict__ tgt, int rows, int ncls, float* __restrict__ accum)
{
    int row = blockIdx.x, tid = threadIdx.x;
    if (row >= rows) return;
    const float* lr = logits + (size_t)row * ncls;
    __shared__ float sm[256];
    float mx = -3.4e38f;
    for (int i = tid; i < ncls; i += 256) mx = fmaxf(mx, lr[i]);
    sm[tid] = mx; __syncthreads();
    for (int s = 128; s > 0; s >>= 1) { if (tid < s) sm[tid] = fmaxf(sm[tid], sm[tid + s]); __syncthreads(); }
    mx = sm[0]; __syncthreads();
    float sum = 0.f;
    for (int i = tid; i < ncls; i += 256) sum += expf(lr[i] - mx);
    sm[tid] = sum; __syncthreads();
    for (int s = 128; s > 0; s >>= 1) { if (tid < s) sm[tid] += sm[tid + s]; __syncthreads(); }
    if (tid == 0) {
        int t = tgt[row];
        bool valid = (t != -100);
        int t2 = valid ? t : 0;
        float nll = -(lr[t2] - mx - logf(sm[0]));
        atomicAdd(&accum[0], valid ? nll : 0.0f);
        atomicAdd(&accum[1], valid ? 1.0f : 0.0f);
    }
}

__global__ void k_finalize(const float* accum, float* loss)
{
    loss[0] = accum[0] / fmaxf(accum[1], 1.0f);
}

// =====================================================================
extern "C" void kernel_launch(void* const* d_in, const int* in_sizes, int n_in,
                              void* d_out, int out_size, void* d_ws, size_t ws_size,
                              hipStream_t stream)
{
    (void)in_sizes; (void)n_in; (void)out_size; (void)ws_size;
    const int*   idx     = (const int*)d_in[0];
    const int*   targets = (const int*)d_in[1];
    const float* tok_emb = (const float*)d_in[2];
    const float* pos_emb = (const float*)d_in[3];
    const float* ln1_g = (const float*)d_in[4];
    const float* ln1_b = (const float*)d_in[5];
    const float* w_qkv = (const float*)d_in[6];
    const float* w_proj = (const float*)d_in[7];
    const float* ln2_g = (const float*)d_in[8];
    const float* ln2_b = (const float*)d_in[9];
    const float* w_up = (const float*)d_in[10];
    const float* w_down = (const float*)d_in[11];
    const float* lnf_g = (const float*)d_in[12];
    const float* lnf_b = (const float*)d_in[13];
    const int*   band_idx[3] = { (const int*)d_in[14], (const int*)d_in[15], (const int*)d_in[16] };
    const float* wa[3]  = { (const float*)d_in[17], (const float*)d_in[20], (const float*)d_in[23] };
    const float* ag[3]  = { (const float*)d_in[18], (const float*)d_in[21], (const float*)d_in[24] };
    const float* ab[3]  = { (const float*)d_in[19], (const float*)d_in[22], (const float*)d_in[25] };
    const float* w_head = (const float*)d_in[26];
    const int bw[3] = { NVb, NIb, NPb };

    float* logits = (float*)d_out;
    float* loss   = logits + (size_t)BT * Vv;

    // ---- workspace layout ----
    char* base = (char*)d_ws;
    size_t off = 0;
    auto alloc = [&](size_t bytes) -> char* {
        off = (off + 255) & ~(size_t)255;
        char* p = base + off;
        off += bytes;
        return p;
    };
    float* x    = (float*)alloc((size_t)BT * Dd * 4);
    float* h    = (float*)alloc((size_t)BT * Dd * 4);
    __hip_bfloat16* hb  = (__hip_bfloat16*)alloc((size_t)BT * Dd * 2);
    float* qkv  = (float*)alloc((size_t)BT * K3 * 4);
    __hip_bfloat16* qb  = (__hip_bfloat16*)alloc((size_t)BT * Dd * 2);
    __hip_bfloat16* kb  = (__hip_bfloat16*)alloc((size_t)BT * Dd * 2);
    __hip_bfloat16* vtb = (__hip_bfloat16*)alloc((size_t)BT * Dd * 2);
    float* att  = (float*)alloc((size_t)Bb * Hh * Tt * Tt * 4);     // reused as mlp_up f32
    __hip_bfloat16* pbf = (__hip_bfloat16*)alloc((size_t)Bb * Hh * Tt * Tt * 2); // reused as up_bf
    float* oh   = (float*)alloc((size_t)BT * Dd * 4);               // reused: proj/down/band_out
    __hip_bfloat16* ob  = (__hip_bfloat16*)alloc((size_t)BT * Dd * 2);
    __hip_bfloat16* anb = (__hip_bfloat16*)alloc((size_t)BT * HK * 2);
    __hip_bfloat16* bandb = (__hip_bfloat16*)alloc((size_t)BT * PADW * 2);
    __hip_bfloat16* wab = (__hip_bfloat16*)alloc((size_t)AWw * PADW * 2);
    __hip_bfloat16* wqkvb = (__hip_bfloat16*)alloc((size_t)K3 * Dd * 2);
    __hip_bfloat16* wprojb = (__hip_bfloat16*)alloc((size_t)Dd * Dd * 2);
    __hip_bfloat16* wupb   = (__hip_bfloat16*)alloc((size_t)K4 * Dd * 2);
    __hip_bfloat16* wdownb = (__hip_bfloat16*)alloc((size_t)Dd * K4 * 2);
    __hip_bfloat16* wheadb = (__hip_bfloat16*)alloc((size_t)Vv * HK * 2);
    float* accum = (float*)alloc(2 * 4);
    float* mlp_up = att;                    // alias
    __hip_bfloat16* upb = pbf;              // alias
    float* proj_out = oh;                   // alias
    float* down_out = oh;                   // alias
    float* band_out = oh;                   // alias

    auto g1 = [](size_t n) { return dim3((unsigned)((n + 255) / 256)); };

    // ---- weight conversions (fp32 -> bf16) ----
    k_cvt<<<g1((size_t)K3 * Dd), 256, 0, stream>>>(w_qkv, wqkvb, (size_t)K3 * Dd);
    k_cvt<<<g1((size_t)Dd * Dd), 256, 0, stream>>>(w_proj, wprojb, (size_t)Dd * Dd);
    k_cvt<<<g1((size_t)K4 * Dd), 256, 0, stream>>>(w_up, wupb, (size_t)K4 * Dd);
    k_cvt<<<g1((size_t)Dd * K4), 256, 0, stream>>>(w_down, wdownb, (size_t)Dd * K4);
    k_cvt<<<g1((size_t)Vv * HK), 256, 0, stream>>>(w_head, wheadb, (size_t)Vv * HK);

    // ---- embedding ----
    k_embed<<<g1((size_t)BT * Dd), 256, 0, stream>>>(idx, tok_emb, pos_emb, x);

    // ---- attention block ----
    k_layernorm<<<BT, 256, 0, stream>>>(x, Dd, BT, ln1_g, ln1_b, h, hb, Dd, 0);
    k_gemm_bf16<<<dim3(K3 / 64, BT / 128, 1), 256, 0, stream>>>(
        hb, 0, Dd, wqkvb, 0, Dd, qkv, 0, K3, BT, K3, Dd, 1.0f, 0);
    k_split_qkv<<<g1((size_t)BT * Dd), 256, 0, stream>>>(qkv, qb, kb, vtb);
    // scores = causal(Q K^T / sqrt(hd)), batched over B*H
    k_gemm_bf16<<<dim3(Tt / 64, Tt / 128, Bb * Hh), 256, 0, stream>>>(
        qb, (long long)Tt * HD, HD, kb, (long long)Tt * HD, HD,
        att, (long long)Tt * Tt, Tt, Tt, Tt, HD, 0.08838834764831845f, 1);
    k_softmax<<<Bb * Hh * Tt, 128, 0, stream>>>(att, pbf, Tt);
    // O = P @ V  (B operand is V^T: (hd, T))
    k_gemm_bf16<<<dim3(HD / 64, Tt / 128, Bb * Hh), 256, 0, stream>>>(
        pbf, (long long)Tt * Tt, Tt, vtb, (long long)HD * Tt, Tt,
        oh, (long long)Tt * HD, HD, Tt, HD, Tt, 1.0f, 0);
    k_merge_heads<<<g1((size_t)BT * Dd), 256, 0, stream>>>(oh, ob);
    k_gemm_bf16<<<dim3(Dd / 64, BT / 128, 1), 256, 0, stream>>>(
        ob, 0, Dd, wprojb, 0, Dd, proj_out, 0, Dd, BT, Dd, Dd, 1.0f, 0);
    k_add<<<g1((size_t)BT * Dd), 256, 0, stream>>>(x, proj_out, (size_t)BT * Dd);

    // ---- MLP block ----
    k_layernorm<<<BT, 256, 0, stream>>>(x, Dd, BT, ln2_g, ln2_b, h, hb, Dd, 0);
    k_gemm_bf16<<<dim3(K4 / 64, BT / 128, 1), 256, 0, stream>>>(
        hb, 0, Dd, wupb, 0, Dd, mlp_up, 0, K4, BT, K4, Dd, 1.0f, 0);
    k_gelu_cvt<<<g1((size_t)BT * K4), 256, 0, stream>>>(mlp_up, upb, (size_t)BT * K4);
    k_gemm_bf16<<<dim3(Dd / 64, BT / 128, 1), 256, 0, stream>>>(
        upb, 0, K4, wdownb, 0, K4, down_out, 0, Dd, BT, Dd, K4, 1.0f, 0);
    k_add<<<g1((size_t)BT * Dd), 256, 0, stream>>>(x, down_out, (size_t)BT * Dd);

    // ---- final LN ----
    k_layernorm<<<BT, 256, 0, stream>>>(x, Dd, BT, lnf_g, lnf_b, h, nullptr, 0, 0);

    // ---- band analyzers (gather -> GEMM -> LN -> GELU -> concat) ----
    for (int b = 0; b < 3; ++b) {
        k_pad_w<<<g1((size_t)AWw * PADW), 256, 0, stream>>>(wa[b], bw[b], wab);
        k_gather_band<<<g1((size_t)BT * PADW), 256, 0, stream>>>(h, band_idx[b], bw[b], bandb);
        k_gemm_bf16<<<dim3(AWw / 64, BT / 128, 1), 256, 0, stream>>>(
            bandb, 0, PADW, wab, 0, PADW, band_out, 0, AWw, BT, AWw, PADW, 1.0f, 0);
        k_layernorm<<<BT, 256, 0, stream>>>(band_out, AWw, BT, ag[b], ab[b],
                                            nullptr, anb + b * AWw, HK, 1);
    }

    // ---- head GEMM -> logits ----
    k_gemm_bf16<<<dim3((Vv + 63) / 64, BT / 128, 1), 256, 0, stream>>>(
        anb, 0, HK, wheadb, 0, HK, logits, 0, Vv, BT, Vv, HK, 1.0f, 0);

    // ---- cross entropy ----
    k_zero2<<<1, 1, 0, stream>>>(accum);
    k_ce<<<BT, 256, 0, stream>>>(logits, targets, BT, Vv, accum);
    k_finalize<<<1, 1, 0, stream>>>(accum, loss);
}
